// HumanVAttention_57354993270964
// MI455X (gfx1250) — matrix operand order repaired and checked by reading, not compiled, not run on hardware
//
#include <hip/hip_runtime.h>
#include <hip/hip_bf16.h>

// ---------------------------------------------------------------------------
// Block-sparse GQA attention layer for MI455X (gfx1250, wave32, WMMA).
// S=4096, H=2048, NH=16, NKV=4, HD=128, B_BLK=64, T=64, MAXSEL=7.
// Pipeline: f32->f16 casts, double-buffered 128x128 WMMA GEMM (QKV + out
// proj), RoPE (scale folded into Q), flash-style block-sparse WMMA attention.
// All LDS tiles use padded row strides to avoid 16-way bank conflicts on
// ds_load_b128 fragment reads.
// ---------------------------------------------------------------------------

typedef __attribute__((ext_vector_type(16))) _Float16 v16h;
typedef __attribute__((ext_vector_type(8)))  _Float16 v8h;   // 16 bytes
typedef __attribute__((ext_vector_type(8)))  float    v8f;

#define S_LEN 4096
#define H_DIM 2048
#define NH    16
#define NKV   4
#define HD    128
#define TBLK  64
#define NEGV  (-1.0e9f)

// Padded LDS row strides (in halves / f32) chosen so (stride*n) mod 64 dwords
// is distinct for n = 0..15 -> conflict-free b128 fragment loads.
#define GLD   40    // GEMM tiles: 32 K-halves + 8 pad (80 B = 20 dwords)
#define QLD   136   // attention Q/K tiles: 128 + 8 (272 B = 68 dwords)
#define VLD   72    // attention V^T tile: 64 + 8 (144 B = 36 dwords)
#define SCLD  68    // score rows: 64 f32 + 4 pad (272 B = 68 dwords)

__device__ __forceinline__ v8f wmma_f16(v16h a, v16h b, v8f c) {
    // v_wmma_f32_16x16x32_f16 : D = A(16x32) x B(32x16) + C
    return __builtin_amdgcn_wmma_f32_16x16x32_f16(
        /*neg_a=*/false, a, /*neg_b=*/false, b,
        /*c_mod=*/(short)0, c, /*reuse_a=*/false, /*reuse_b=*/false);
}

// A-fragment 16x32 f16 from row-major LDS tile (row stride ld halves).
// lane m = lane&15; lanes 0-15: K 0-7 & 16-23; lanes 16-31: K 8-15 & 24-31.
__device__ __forceinline__ v16h frag_a(const _Float16* p, int ld) {
    int lane = threadIdx.x & 31;
    int m = lane & 15, kh = lane >> 4;
    const _Float16* r = p + m * ld;
    v16h a;
    *(v8h*)&a       = *(const v8h*)(r + kh * 8);
    *((v8h*)&a + 1) = *(const v8h*)(r + 16 + kh * 8);
    return a;
}

// B-fragment 32x16 f16 from an LDS tile stored [N][K] row-major.
// lane n = lane&15; lanes 0-15: K 0-7 & 16-23; lanes 16-31: K 8-15 & 24-31.
__device__ __forceinline__ v16h frag_b(const _Float16* p, int ld) {
    int lane = threadIdx.x & 31;
    int n = lane & 15, kh = lane >> 4;
    const _Float16* r = p + n * ld + kh * 8;
    v16h b;
    *(v8h*)&b       = *(const v8h*)(r);
    *((v8h*)&b + 1) = *(const v8h*)(r + 16);
    return b;
}

// ---------------------------------------------------------------------------
__global__ void cvt_f32_f16(const float* __restrict__ in,
                            _Float16* __restrict__ out, int n) {
    int i = blockIdx.x * 256 + threadIdx.x;
    if (i < n) out[i] = (_Float16)in[i];
}

// ---------------------------------------------------------------------------
// C[M,N] = A[M,K] * B[N,K]^T   (A,B f16; C f16 or f32)
// Block tile 128x128, double-buffered LDS (K-step 32, one barrier/step).
// 8 waves: wave w -> 32x64 region: m-tiles {2*(w&3), +1}, n-tiles 4*(w>>2)+j.
// 8 WMMAs per wave per K-step.
// ---------------------------------------------------------------------------
__global__ __launch_bounds__(256) void gemm_wmma(
    const _Float16* __restrict__ A, const _Float16* __restrict__ B,
    _Float16* __restrict__ Ch, float* __restrict__ Cf,
    int M, int N, int K) {
    __shared__ _Float16 As[2][128 * GLD];
    __shared__ _Float16 Bs[2][128 * GLD];
    int n0 = blockIdx.x * 128, m0 = blockIdx.y * 128;
    int tid = threadIdx.x, w = tid >> 5, lane = tid & 31;
    int wm = w & 3;          // m-block: rows wm*32 .. +31
    int wn = w >> 2;         // n-block: cols wn*64 .. +63
    v8f acc[2][4] = {};

    auto load_tile = [&](int buf, int k0) {
#pragma unroll
        for (int r = 0; r < 2; ++r) {
            int idx = tid + r * 256;            // 512 chunks of 8 halves
            int row = idx >> 2, col = (idx & 3) * 8;
            *(v8h*)&As[buf][row * GLD + col] =
                *(const v8h*)(A + (size_t)(m0 + row) * K + k0 + col);
            *(v8h*)&Bs[buf][row * GLD + col] =
                *(const v8h*)(B + (size_t)(n0 + row) * K + k0 + col);
        }
    };

    load_tile(0, 0);
    int nk = K >> 5;
    for (int ki = 0; ki < nk; ++ki) {
        __syncthreads();                        // buf (ki&1) ready; reads of it
        if (ki + 1 < nk) load_tile((ki + 1) & 1, (ki + 1) * 32);
        const _Float16* ab = &As[ki & 1][0];
        const _Float16* bb = &Bs[ki & 1][0];
        v16h a0 = frag_a(ab + (wm * 2 + 0) * 16 * GLD, GLD);
        v16h a1 = frag_a(ab + (wm * 2 + 1) * 16 * GLD, GLD);
#pragma unroll
        for (int j = 0; j < 4; ++j) {
            v16h b = frag_b(bb + (wn * 4 + j) * 16 * GLD, GLD);
            acc[0][j] = wmma_f16(a0, b, acc[0][j]);
            acc[1][j] = wmma_f16(a1, b, acc[1][j]);
        }
    }

    int n = lane & 15, mo = lane >> 4;
#pragma unroll
    for (int i = 0; i < 2; ++i) {
#pragma unroll
        for (int r = 0; r < 8; ++r) {
            int m = m0 + (wm * 2 + i) * 16 + r + 8 * mo;
#pragma unroll
            for (int j = 0; j < 4; ++j) {
                int c = n0 + (wn * 4 + j) * 16 + n;
                if (Cf) Cf[(size_t)m * N + c] = acc[i][j][r];
                else    Ch[(size_t)m * N + c] = (_Float16)acc[i][j][r];
            }
        }
    }
}

// ---------------------------------------------------------------------------
// In-place RoPE on x[S][nh*128]; one thread per (s,h,d<64) pair -> race free.
// out[d]    =  x[d]*cos[d]       - x[d+64]*sin[d]
// out[d+64] =  x[d+64]*cos[d+64] + x[d]*sin[d+64]
// scale folds the attention 1/sqrt(HD) into Q.
// ---------------------------------------------------------------------------
__global__ void rope_kernel(_Float16* __restrict__ x,
                            const float* __restrict__ cosb,
                            const float* __restrict__ sinb,
                            int nh, float scale) {
    int idx = blockIdx.x * 256 + threadIdx.x;
    int total = S_LEN * nh * 64;
    if (idx >= total) return;
    int d = idx & 63;
    int h = (idx >> 6) % nh;
    int s = idx / (nh * 64);
    float c0 = cosb[s * HD + d], c1 = cosb[s * HD + d + 64];
    float s0 = sinb[s * HD + d], s1 = sinb[s * HD + d + 64];
    _Float16* p = x + (size_t)s * (nh * HD) + h * HD + d;
    float x0 = (float)p[0], x1 = (float)p[64];
    p[0]  = (_Float16)((x0 * c0 - x1 * s0) * scale);
    p[64] = (_Float16)((x1 * c1 + x0 * s1) * scale);
}

// ---------------------------------------------------------------------------
// Flash-style block-sparse attention. One block per (head, q-block).
// 8 waves: wave w -> score tiles (mt=w&3, nt {2*(w>>2),+1});
//          output tiles (mt=w&3, d-half dh=w>>2, 4 n-tiles of 16).
// LDS: qs (padded) | kv (K tile, then V^T tile) | sc (f32 scores, front half
//      of each row reused as f16 probs) | row state.  ~54 KB total.
// ---------------------------------------------------------------------------
__global__ __launch_bounds__(256) void attn_kernel(
    const _Float16* __restrict__ Q,   // [S][NH*HD], RoPE'd, pre-scaled
    const _Float16* __restrict__ Kc,  // [S][NKV*HD], RoPE'd
    const _Float16* __restrict__ Vc,  // [S][NKV*HD]
    _Float16* __restrict__ O) {       // [S][NH*HD]
    __shared__ _Float16 qs[TBLK * QLD];
    __shared__ _Float16 kv[HD * VLD];       // 9216 halves >= 64*QLD (8704)
    __shared__ float    sc[TBLK * SCLD];
    __shared__ float    rm[TBLK], rl[TBLK], rsc[TBLK];

    int t = blockIdx.x & 63;
    int h = blockIdx.x >> 6;
    int kvh = h >> 2;                       // GQA: NH/NKV = 4
    int tid = threadIdx.x, w = tid >> 5, lane = tid & 31;

    // Static block-selection mask (LOCAL=4 + diag + GLOB=2 of stride-4 grid).
    unsigned long long msel = 0ull;
    for (int i = (t > 4 ? t - 4 : 0); i <= t; ++i) msel |= 1ull << i;
    int ncand = t / 4 + 1;
    msel |= 1ull << ((ncand - 1) * 4);
    if (ncand >= 2) msel |= 1ull << ((ncand - 2) * 4);

    // Stage Q tile (64x128 f16, already scaled).
#pragma unroll
    for (int r = 0; r < 4; ++r) {
        int idx = tid + r * 256;            // 1024 chunks of 8 halves
        int row = idx >> 4, col = (idx & 15) * 8;
        *(v8h*)&qs[row * QLD + col] =
            *(const v8h*)(Q + (size_t)(t * TBLK + row) * (NH * HD) + h * HD + col);
    }
    if (tid < TBLK) { rm[tid] = -3.0e38f; rl[tid] = 0.0f; }

    int mt = w & 3, dh = w >> 2;
    v8f acc[4] = {{}, {}, {}, {}};
    __syncthreads();

    for (int kb = 0; kb < 64; ++kb) {
        if (!((msel >> kb) & 1ull)) continue;   // uniform across block
        int s0 = kb * TBLK;

        __syncthreads();                        // prev PV done before kv reuse
        // Stage K tile [64][128] (padded rows)
#pragma unroll
        for (int r = 0; r < 4; ++r) {
            int idx = tid + r * 256;
            int row = idx >> 4, col = (idx & 15) * 8;
            *(v8h*)&kv[row * QLD + col] =
                *(const v8h*)(Kc + (size_t)(s0 + row) * (NKV * HD) + kvh * HD + col);
        }
        __syncthreads();

        // Scores: S = Q(64x128) * K^T(128x64), wave -> 2 of 16 tiles
        {
            int nt0 = (w >> 2) * 2, nt1 = nt0 + 1;
            v8f sv0 = {}, sv1 = {};
#pragma unroll
            for (int kk = 0; kk < HD; kk += 32) {
                v16h a  = frag_a(qs + mt * 16 * QLD + kk, QLD);
                v16h b0 = frag_b(kv + nt0 * 16 * QLD + kk, QLD);
                v16h b1 = frag_b(kv + nt1 * 16 * QLD + kk, QLD);
                sv0 = wmma_f16(a, b0, sv0);
                sv1 = wmma_f16(a, b1, sv1);
            }
            bool diag = (kb == t);
            int n = lane & 15, mo = lane >> 4;
#pragma unroll
            for (int r = 0; r < 8; ++r) {
                int m  = mt * 16 + r + 8 * mo;
                int c0 = nt0 * 16 + n, c1 = c0 + 16;
                float x0 = sv0[r], x1 = sv1[r];
                if (diag && c0 > m) x0 = NEGV;
                if (diag && c1 > m) x1 = NEGV;
                sc[m * SCLD + c0] = x0;
                sc[m * SCLD + c1] = x1;
            }
        }
        __syncthreads();                        // scores done; K tile free

        // Stage V transposed [128][64] into kv (B-frag wants contiguous K=j).
#pragma unroll
        for (int r = 0; r < 4; ++r) {
            int idx = tid + r * 256;
            int j = idx >> 4, cd = (idx & 15) * 8;
            v8h vv = *(const v8h*)(Vc + (size_t)(s0 + j) * (NKV * HD) + kvh * HD + cd);
#pragma unroll
            for (int e = 0; e < 8; ++e) kv[(cd + e) * VLD + j] = vv[e];
        }

        // Online softmax: wave w owns rows w*8 .. w*8+7 (64 cols, 2/lane).
        for (int q8 = 0; q8 < 8; ++q8) {
            int row = w * 8 + q8;
            float e0 = sc[row * SCLD + lane];
            float e1 = sc[row * SCLD + 32 + lane];
            float mx = fmaxf(e0, e1);
#pragma unroll
            for (int off = 16; off; off >>= 1)
                mx = fmaxf(mx, __shfl_xor(mx, off, 32));
            float mnew = fmaxf(rm[row], mx);
            e0 = __expf(e0 - mnew);
            e1 = __expf(e1 - mnew);
            float sum = e0 + e1;
#pragma unroll
            for (int off = 16; off; off >>= 1)
                sum += __shfl_xor(sum, off, 32);
            // f16 probs aliased into the front 128 B of this row's f32 slot
            _Float16* pr = (_Float16*)(sc + row * SCLD);
            pr[lane]      = (_Float16)e0;
            pr[lane + 32] = (_Float16)e1;
            if (lane == 0) {
                float f = __expf(rm[row] - mnew);
                rsc[row] = f;
                rl[row]  = rl[row] * f + sum;
                rm[row]  = mnew;
            }
        }
        __syncthreads();                        // probs, V^T, rescale ready

        // Rescale accumulators, then O += P(64x64) * V(64x128)
        {
            int mo = lane >> 4;
#pragma unroll
            for (int r = 0; r < 8; ++r) {
                float f = rsc[mt * 16 + r + 8 * mo];
#pragma unroll
                for (int ntl = 0; ntl < 4; ++ntl) acc[ntl][r] *= f;
            }
            const _Float16* prb = (const _Float16*)sc;   // row stride 2*SCLD halves
#pragma unroll
            for (int kk = 0; kk < TBLK; kk += 32) {
                v16h a = frag_a(prb + mt * 16 * (2 * SCLD) + kk, 2 * SCLD);
#pragma unroll
                for (int ntl = 0; ntl < 4; ++ntl) {
                    v16h b = frag_b(kv + (dh * 64 + ntl * 16) * VLD + kk, VLD);
                    acc[ntl] = wmma_f16(a, b, acc[ntl]);
                }
            }
        }
    }
    __syncthreads();

    // Normalize and write O tile.
    {
        int n = lane & 15, mo = lane >> 4;
#pragma unroll
        for (int r = 0; r < 8; ++r) {
            int mrow = mt * 16 + r + 8 * mo;
            float inv = 1.0f / rl[mrow];
            size_t gs = (size_t)(t * TBLK + mrow) * (NH * HD) + h * HD;
#pragma unroll
            for (int ntl = 0; ntl < 4; ++ntl) {
                int d = dh * 64 + ntl * 16 + n;
                O[gs + d] = (_Float16)(acc[ntl][r] * inv);
            }
        }
    }
}

// ---------------------------------------------------------------------------
extern "C" void kernel_launch(void* const* d_in, const int* in_sizes, int n_in,
                              void* d_out, int out_size, void* d_ws, size_t ws_size,
                              hipStream_t stream) {
    (void)in_sizes; (void)n_in; (void)out_size; (void)ws_size;
    const float* hs   = (const float*)d_in[0];
    const float* cosb = (const float*)d_in[1];
    const float* sinb = (const float*)d_in[2];
    const float* wq   = (const float*)d_in[3];
    const float* wk   = (const float*)d_in[4];
    const float* wv   = (const float*)d_in[5];
    const float* wo   = (const float*)d_in[6];
    float* out = (float*)d_out;

    char* ws = (char*)d_ws;
    size_t off = 0;
    _Float16* hs16 = (_Float16*)(ws + off); off += (size_t)S_LEN * H_DIM * 2;          // 16 MB
    _Float16* wq16 = (_Float16*)(ws + off); off += (size_t)H_DIM * H_DIM * 2;          //  8 MB
    _Float16* wk16 = (_Float16*)(ws + off); off += (size_t)(NKV * HD) * H_DIM * 2;     //  2 MB
    _Float16* wv16 = (_Float16*)(ws + off); off += (size_t)(NKV * HD) * H_DIM * 2;     //  2 MB
    _Float16* wo16 = (_Float16*)(ws + off); off += (size_t)H_DIM * H_DIM * 2;          //  8 MB
    _Float16* q16  = (_Float16*)(ws + off); off += (size_t)S_LEN * (NH * HD) * 2;      // 16 MB
    _Float16* k16  = (_Float16*)(ws + off); off += (size_t)S_LEN * (NKV * HD) * 2;     //  4 MB
    _Float16* v16  = (_Float16*)(ws + off); off += (size_t)S_LEN * (NKV * HD) * 2;     //  4 MB
    _Float16* ao16 = (_Float16*)(ws + off); off += (size_t)S_LEN * (NH * HD) * 2;      // 16 MB

    // f32 -> f16 casts
    {
        int n;
        n = S_LEN * H_DIM;      cvt_f32_f16<<<(n + 255) / 256, 256, 0, stream>>>(hs, hs16, n);
        n = H_DIM * H_DIM;      cvt_f32_f16<<<(n + 255) / 256, 256, 0, stream>>>(wq, wq16, n);
        n = NKV * HD * H_DIM;   cvt_f32_f16<<<(n + 255) / 256, 256, 0, stream>>>(wk, wk16, n);
        n = NKV * HD * H_DIM;   cvt_f32_f16<<<(n + 255) / 256, 256, 0, stream>>>(wv, wv16, n);
        n = H_DIM * H_DIM;      cvt_f32_f16<<<(n + 255) / 256, 256, 0, stream>>>(wo, wo16, n);
    }

    // QKV projections (WMMA GEMM, f16 out), 128x128 block tiles
    gemm_wmma<<<dim3(H_DIM / 128, S_LEN / 128), 256, 0, stream>>>(
        hs16, wq16, q16, nullptr, S_LEN, H_DIM, H_DIM);
    gemm_wmma<<<dim3((NKV * HD) / 128, S_LEN / 128), 256, 0, stream>>>(
        hs16, wk16, k16, nullptr, S_LEN, NKV * HD, H_DIM);
    gemm_wmma<<<dim3((NKV * HD) / 128, S_LEN / 128), 256, 0, stream>>>(
        hs16, wv16, v16, nullptr, S_LEN, NKV * HD, H_DIM);

    // RoPE (in place); fold 1/sqrt(HD) into Q
    const float SCALE = 0.08838834764831845f;   // 128^-0.5
    {
        int n = S_LEN * NH * 64;
        rope_kernel<<<(n + 255) / 256, 256, 0, stream>>>(q16, cosb, sinb, NH, SCALE);
        n = S_LEN * NKV * 64;
        rope_kernel<<<(n + 255) / 256, 256, 0, stream>>>(k16, cosb, sinb, NKV, 1.0f);
    }

    // Block-sparse flash attention: grid = NH * T = 1024 blocks
    attn_kernel<<<NH * (S_LEN / TBLK), 256, 0, stream>>>(q16, k16, v16, ao16);

    // Output projection -> f32 d_out
    gemm_wmma<<<dim3(H_DIM / 128, S_LEN / 128), 256, 0, stream>>>(
        ao16, wo16, nullptr, out, S_LEN, H_DIM, H_DIM);
}
